// MultiheadSelfAttention_5050881540677
// MI455X (gfx1250) — matrix operand order repaired and checked by reading, not compiled, not run on hardware
//
#include <hip/hip_runtime.h>
#include <hip/hip_bf16.h>

// ---------------- problem constants ----------------
constexpr int NB = 4;          // batch
constexpr int NL = 2048;       // seq len
constexpr int NC = 512;        // channels
constexpr int NH = 8;          // heads
constexpr int ND = 64;         // head dim
constexpr int NBL = NB * NL;   // 8192 rows
constexpr int NQKV = 3 * NC;   // 1536

typedef __bf16 bf16_t;
typedef __attribute__((ext_vector_type(16))) __bf16 v16bf;
typedef __attribute__((ext_vector_type(8)))  float  v8f;

union FragU { v16bf v; uint4 q[2]; };

__device__ __forceinline__ bf16_t f2bf(float f) {
  union { float f; unsigned u; } x; x.f = f;
  unsigned r = x.u + 0x7FFFu + ((x.u >> 16) & 1u);   // round-to-nearest-even
  unsigned short hb = (unsigned short)(r >> 16);
  union { unsigned short s; bf16_t b; } y; y.s = hb;
  return y.b;
}

// A operand 16x32 bf16: lane = row M; lanes0-15 K0..7(v0-3),K16..23(v4-7); lanes16-31 +8
__device__ __forceinline__ v16bf load_a16(const bf16_t* base, int ld, int row0, int col0) {
  const int lane = threadIdx.x & 31;
  const int sub  = lane >> 4;                 // 0 or 1
  const bf16_t* p = base + (size_t)(row0 + (lane & 15)) * ld + col0 + sub * 8;
  FragU u;
  u.q[0] = *(const uint4*)p;
  u.q[1] = *(const uint4*)(p + 16);
  return u.v;
}

// B operand 32x16 bf16: lane = col N; lanes0-15 K0..7,K16..23; lanes16-31 +8
__device__ __forceinline__ v16bf load_b16(const bf16_t* base, int ld, int row0, int col0) {
  const int lane = threadIdx.x & 31;
  const int sub  = lane >> 4;
  const bf16_t* p = base + (size_t)(row0 + (lane & 15)) * ld + col0 + sub * 8;
  FragU u;
  u.q[0] = *(const uint4*)p;
  u.q[1] = *(const uint4*)(p + 16);
  return u.v;
}

__device__ __forceinline__ v8f wmma_bf(v16bf a, v16bf b, v8f c) {
  return __builtin_amdgcn_wmma_f32_16x16x32_bf16(false, a, false, b, (short)0, c,
                                                 false, false);
}

// gfx1250 async global->LDS copy (ASYNCcnt-tracked), 16B per lane
__device__ __forceinline__ void async_ld_b128(const void* gptr, const void* lptr) {
  unsigned long long ga = (unsigned long long)(uintptr_t)gptr;
  unsigned int la = (unsigned int)(uintptr_t)lptr;   // low 32 bits = LDS byte offset
  asm volatile("global_load_async_to_lds_b128 %0, %1, off"
               :: "v"(la), "v"(ga) : "memory");
}
__device__ __forceinline__ void wait_async0() {
  asm volatile("s_wait_asynccnt 0x0" ::: "memory");
}

// ---------------- fp32 -> bf16 convert ----------------
__global__ void convert_f32_bf16(const float* __restrict__ in, bf16_t* __restrict__ out, int n) {
  int i = blockIdx.x * blockDim.x + threadIdx.x;
  if (i < n) out[i] = f2bf(in[i]);
}

// ---------------- QKV GEMM + bias + RoPE + scatter ----------------
// grid: (NBL/128, NQKV/64), block 256 (8 waves). Wave: 16 rows x 64 cols, K=512.
__global__ __launch_bounds__(256) void qkv_rope_kernel(
    const bf16_t* __restrict__ xb, const bf16_t* __restrict__ wb,
    const float* __restrict__ qkv_b,
    bf16_t* __restrict__ qbuf, bf16_t* __restrict__ kbuf, bf16_t* __restrict__ vt) {
  const int lane = threadIdx.x & 31;
  const int wave = threadIdx.x >> 5;
  const int sub  = lane >> 4;
  const int row0 = blockIdx.x * 128 + wave * 16;
  const int col0 = blockIdx.y * 64;

  v8f z = 0.0f;
  v8f acc[4] = {z, z, z, z};
  for (int k0 = 0; k0 < NC; k0 += 32) {
    v16bf a = load_a16(xb, NC, row0, k0);
#pragma unroll
    for (int f = 0; f < 4; ++f)
      acc[f] = wmma_bf(a, load_b16(wb, NC, col0 + 16 * f, k0), acc[f]);
  }
#pragma unroll
  for (int f = 0; f < 4; ++f) {
    float bias = qkv_b[col0 + 16 * f + (lane & 15)];
#pragma unroll
    for (int j = 0; j < 8; ++j) acc[f][j] += bias;
  }

  const int sec = col0 / NC;           // 0=q, 1=k, 2=v (64-col tile lies in one section)

  if (sec < 2) {
    // RoPE: pair fragments (0,1) and (2,3): channels d and d+16 of a 32-group,
    // same lane in adjacent accumulators. inv_freq[j] = 10000^(-j/32).
    const int d0 = lane & 15;
    const float if0 = __expf(-0.28782313662425572f * (float)d0);
    const float if1 = __expf(-0.28782313662425572f * (float)(d0 + 16));
#pragma unroll
    for (int pp = 0; pp < 2; ++pp) {
#pragma unroll
      for (int j = 0; j < 8; ++j) {
        int grow = row0 + j + sub * 8;
        float pos = (float)(grow & (NL - 1));
        float a0 = pos * if0, a1 = pos * if1;
        float c0 = __cosf(a0), s0 = __sinf(a0);
        float c1 = __cosf(a1), s1 = __sinf(a1);
        float x0 = acc[2 * pp + 0][j], x1 = acc[2 * pp + 1][j];
        acc[2 * pp + 0][j] = x0 * c0 - x1 * s0;   // d < 16: rot = -x[d+16]
        acc[2 * pp + 1][j] = x1 * c1 + x0 * s1;   // d >=16: rot = +x[d-16]
      }
    }
  }

  // Scatter: q,k -> [b][h][L][64] row-major; v -> transposed [b][h][64][L]
#pragma unroll
  for (int f = 0; f < 4; ++f) {
    int n  = col0 + 16 * f + (lane & 15);
    int c  = n & (NC - 1);
    int h  = c >> 6;
    int hd = c & 63;
#pragma unroll
    for (int j = 0; j < 8; ++j) {
      int grow = row0 + j + sub * 8;
      int bi = grow >> 11;
      int l  = grow & (NL - 1);
      bf16_t val = f2bf(acc[f][j]);
      size_t bh = (size_t)(bi * NH + h);
      if (sec == 0)      qbuf[(bh * NL + l) * ND + hd] = val;
      else if (sec == 1) kbuf[(bh * NL + l) * ND + hd] = val;
      else               vt[(bh * ND + hd) * NL + l]   = val;
    }
  }
}

// ---------------- flash attention ----------------
// One block = 128 queries of one (b,h); 8 waves x 16 queries. K/V tiles for each
// 32-key step are staged in LDS with double-buffered async global->LDS copies so
// every wave reads them via ds_load instead of re-fetching from global (8x less
// global K/V traffic).
__global__ __launch_bounds__(256) void attn_kernel(
    const bf16_t* __restrict__ qbuf, const bf16_t* __restrict__ kbuf,
    const bf16_t* __restrict__ vt, bf16_t* __restrict__ attnb) {
  __shared__ __align__(16) bf16_t ktile[2][32 * 64];   // [key][hd]   4KB each
  __shared__ __align__(16) bf16_t vtile[2][64 * 32];   // [hd][key]   4KB each
  __shared__ __align__(16) bf16_t ptile[8][16 * 32];   // per-wave P  1KB each

  const int tid  = threadIdx.x;
  const int lane = tid & 31;
  const int wave = tid >> 5;
  const int sub  = lane >> 4;

  const int blk  = blockIdx.x;               // NB*NH*16 = 512 blocks
  const int qblk = blk & 15;
  const int h    = (blk >> 4) & (NH - 1);
  const int bi   = blk >> 7;
  const size_t bh = (size_t)(bi * NH + h);
  const bf16_t* qp = qbuf + bh * NL * ND;
  const bf16_t* kp = kbuf + bh * NL * ND;
  const bf16_t* vp = vt   + bh * ND * NL;
  bf16_t* pl = ptile[wave];

  const int q0 = qblk * 128 + wave * 16;
  const v16bf qa0 = load_a16(qp, ND, q0, 0);
  const v16bf qa1 = load_a16(qp, ND, q0, 32);

  // cooperative tile-load coordinates (256 threads, 16B each, 4KB per tile)
  const int krow = tid >> 3, kcol = (tid & 7) * 8;   // K: 32 rows x 64 cols
  const int vrow = tid >> 2, vcol = (tid & 3) * 8;   // V: 64 rows x 32 cols

  auto issue_tiles = [&](int buf, int kbeg) {
    async_ld_b128(kp + (size_t)(kbeg + krow) * ND + kcol, &ktile[buf][krow * ND + kcol]);
    async_ld_b128(vp + (size_t)vrow * NL + kbeg + vcol,   &vtile[buf][vrow * 32 + vcol]);
  };

  v8f z = 0.0f;
  v8f o[4] = {z, z, z, z};
  float rm[8], rs[8];
#pragma unroll
  for (int j = 0; j < 8; ++j) { rm[j] = -3.0e38f; rs[j] = 0.0f; }

  issue_tiles(0, 0);
  wait_async0();
  __syncthreads();

  for (int it = 0; it < NL / 32; ++it) {
    const int cur = it & 1;
    if (it + 1 < NL / 32) issue_tiles(cur ^ 1, (it + 1) * 32);

    const bf16_t* K = ktile[cur];
    const bf16_t* V = vtile[cur];

    // scores: two 16x16 tiles over this 32-key block, K-dim 64 in two steps
    v8f s0 = wmma_bf(qa0, load_b16(K, ND, 0, 0), z);
    s0     = wmma_bf(qa1, load_b16(K, ND, 0, 32), s0);
    v8f s1 = wmma_bf(qa0, load_b16(K, ND, 16, 0), z);
    s1     = wmma_bf(qa1, load_b16(K, ND, 16, 32), s1);

    float tmax[8], p0[8], p1[8], corr[8], tsum[8];
#pragma unroll
    for (int j = 0; j < 8; ++j) {
      s0[j] *= 0.125f;  s1[j] *= 0.125f;      // 1/sqrt(64)
      tmax[j] = fmaxf(s0[j], s1[j]);
    }
#pragma unroll
    for (int m = 1; m < 16; m <<= 1)
#pragma unroll
      for (int j = 0; j < 8; ++j)
        tmax[j] = fmaxf(tmax[j], __shfl_xor(tmax[j], m, 32));
#pragma unroll
    for (int j = 0; j < 8; ++j) {
      float mn = fmaxf(rm[j], tmax[j]);
      corr[j] = __expf(rm[j] - mn);
      rm[j] = mn;
      p0[j] = __expf(s0[j] - mn);
      p1[j] = __expf(s1[j] - mn);
      tsum[j] = p0[j] + p1[j];
    }
#pragma unroll
    for (int m = 1; m < 16; m <<= 1)
#pragma unroll
      for (int j = 0; j < 8; ++j)
        tsum[j] += __shfl_xor(tsum[j], m, 32);
#pragma unroll
    for (int j = 0; j < 8; ++j) rs[j] = rs[j] * corr[j] + tsum[j];
#pragma unroll
    for (int f = 0; f < 4; ++f)
#pragma unroll
      for (int j = 0; j < 8; ++j) o[f][j] *= corr[j];

    // D-layout P -> LDS -> A-layout (per-wave private tile, DS kept in order)
    const int colA = lane & 15;
#pragma unroll
    for (int j = 0; j < 8; ++j) {
      int r = j + sub * 8;
      pl[r * 32 + colA]      = f2bf(p0[j]);
      pl[r * 32 + colA + 16] = f2bf(p1[j]);
    }
    v16bf pa;
    {
      const bf16_t* pr = pl + (size_t)(lane & 15) * 32 + sub * 8;
      FragU u;
      u.q[0] = *(const uint4*)pr;
      u.q[1] = *(const uint4*)(pr + 16);
      pa = u.v;
    }
    // O += P(16x32) x V(32x64), V tile read from LDS
#pragma unroll
    for (int f = 0; f < 4; ++f)
      o[f] = wmma_bf(pa, load_b16(V, 32, f * 16, 0), o[f]);

    wait_async0();      // next tiles resident
    __syncthreads();    // all waves done with cur, next buffer visible to all
  }

#pragma unroll
  for (int j = 0; j < 8; ++j) rs[j] = 1.0f / rs[j];
#pragma unroll
  for (int f = 0; f < 4; ++f) {
    int colc = h * ND + f * 16 + (lane & 15);
#pragma unroll
    for (int j = 0; j < 8; ++j) {
      int grow = bi * NL + q0 + j + sub * 8;
      attnb[(size_t)grow * NC + colc] = f2bf(o[f][j] * rs[j]);
    }
  }
}

// ---------------- output projection + bias -> fp32 ----------------
__global__ __launch_bounds__(256) void outproj_kernel(
    const bf16_t* __restrict__ ab, const bf16_t* __restrict__ owb,
    const float* __restrict__ out_b, float* __restrict__ out) {
  const int lane = threadIdx.x & 31;
  const int wave = threadIdx.x >> 5;
  const int sub  = lane >> 4;
  const int row0 = blockIdx.x * 128 + wave * 16;
  const int col0 = blockIdx.y * 64;

  v8f z = 0.0f;
  v8f acc[4] = {z, z, z, z};
  for (int k0 = 0; k0 < NC; k0 += 32) {
    v16bf a = load_a16(ab, NC, row0, k0);
#pragma unroll
    for (int f = 0; f < 4; ++f)
      acc[f] = wmma_bf(a, load_b16(owb, NC, col0 + 16 * f, k0), acc[f]);
  }
#pragma unroll
  for (int f = 0; f < 4; ++f) {
    int n = col0 + 16 * f + (lane & 15);
    float bias = out_b[n];
#pragma unroll
    for (int j = 0; j < 8; ++j) {
      int grow = row0 + j + sub * 8;
      out[(size_t)grow * NC + n] = acc[f][j] + bias;
    }
  }
}

// ---------------- host ----------------
extern "C" void kernel_launch(void* const* d_in, const int* in_sizes, int n_in,
                              void* d_out, int out_size, void* d_ws, size_t ws_size,
                              hipStream_t stream) {
  const float* x     = (const float*)d_in[0];
  const float* qkv_w = (const float*)d_in[1];
  const float* qkv_b = (const float*)d_in[2];
  const float* out_w = (const float*)d_in[3];
  const float* out_b = (const float*)d_in[4];
  float* out = (float*)d_out;

  char* ws = (char*)d_ws;
  size_t off = 0;
  auto walloc = [&](size_t bytes) -> void* {
    void* p = ws + off;
    off += (bytes + 255) & ~(size_t)255;
    return p;
  };
  bf16_t* xb   = (bf16_t*)walloc((size_t)NBL * NC * 2);    // x in bf16
  bf16_t* wb   = (bf16_t*)walloc((size_t)NQKV * NC * 2);   // qkv_w in bf16
  bf16_t* owb  = (bf16_t*)walloc((size_t)NC * NC * 2);     // out_w in bf16
  bf16_t* qbuf = (bf16_t*)walloc((size_t)NBL * NC * 2);    // [b][h][L][64] roped
  bf16_t* kbuf = (bf16_t*)walloc((size_t)NBL * NC * 2);    // [b][h][L][64] roped
  bf16_t* vt   = (bf16_t*)walloc((size_t)NBL * NC * 2);    // [b][h][64][L]
  bf16_t* ab   = (bf16_t*)walloc((size_t)NBL * NC * 2);    // attn out bf16

  int n1 = NBL * NC;
  convert_f32_bf16<<<(n1 + 255) / 256, 256, 0, stream>>>(x, xb, n1);
  int n2 = NQKV * NC;
  convert_f32_bf16<<<(n2 + 255) / 256, 256, 0, stream>>>(qkv_w, wb, n2);
  int n3 = NC * NC;
  convert_f32_bf16<<<(n3 + 255) / 256, 256, 0, stream>>>(out_w, owb, n3);

  qkv_rope_kernel<<<dim3(NBL / 128, NQKV / 64), 256, 0, stream>>>(
      xb, wb, qkv_b, qbuf, kbuf, vt);

  attn_kernel<<<dim3(NB * NH * 16), 256, 0, stream>>>(qbuf, kbuf, vt, ab);

  outproj_kernel<<<dim3(NBL / 128, NC / 64), 256, 0, stream>>>(
      ab, owb, out_b, out);
}
